// GCNet_9028021256690
// MI455X (gfx1250) — compile-verified, matrix operand order checked by reading, not run on hardware
//
#include <hip/hip_runtime.h>
#include <hip/hip_bf16.h>
#include <math.h>

typedef __bf16 bf16;
typedef __attribute__((ext_vector_type(16))) __bf16 v16bf;
typedef __attribute__((ext_vector_type(8)))  float  v8f;
typedef __attribute__((ext_vector_type(4)))  int    v4i;

#define BM 128
#define BN 128
#define BK 32

#if __has_builtin(__builtin_amdgcn_global_load_async_to_lds_b128)
#define GCN_ASYNC 1
#endif

union Frag { v16bf v; uint4 q[2]; };

__device__ __forceinline__ void wait_async0() {
#if __has_builtin(__builtin_amdgcn_s_wait_asynccnt)
  __builtin_amdgcn_s_wait_asynccnt(0);
#else
  asm volatile("s_wait_asynccnt 0x0" ::: "memory");
#endif
}

#ifdef GCN_ASYNC
// Per-lane async copy of 16 bytes global -> LDS (ASYNCcnt-tracked).
__device__ __forceinline__ void async_cp16(const bf16* g, bf16* l) {
  v4i* gg = (v4i*)g;  // strip const + reinterpret (generic AS)
  v4i* ll = (v4i*)l;
  __builtin_amdgcn_global_load_async_to_lds_b128(
      (__attribute__((address_space(1))) v4i*)gg,
      (__attribute__((address_space(3))) v4i*)ll, 0, 0);
}
#endif

// Split fp32 row-major M x K into bf16 hi/lo, zero-padded to M x Kp (Kp % 32 == 0).
__global__ __launch_bounds__(256) void cvt_split_rowmajor(
    const float* __restrict__ src, bf16* __restrict__ hi, bf16* __restrict__ lo,
    int M, int K, int Kp)
{
  int idx = blockIdx.x * 256 + threadIdx.x;
  int total = M * Kp;
  if (idx >= total) return;
  int r = idx / Kp, c = idx - r * Kp;
  float v = (c < K) ? src[(size_t)r * K + c] : 0.0f;
  bf16 h = (bf16)v;
  hi[idx] = h;
  lo[idx] = (bf16)(v - (float)h);
}

// Split fp32 row-major K x N into transposed bf16 hi/lo N x Kp (K-major "BT" layout).
__global__ __launch_bounds__(256) void cvt_split_transpose(
    const float* __restrict__ src, bf16* __restrict__ hi, bf16* __restrict__ lo,
    int K, int N, int Kp)
{
  int idx = blockIdx.x * 256 + threadIdx.x;
  int total = N * Kp;
  if (idx >= total) return;
  int n = idx / Kp, c = idx - n * Kp;
  float v = 0.0f;
  if (c < K) {
    v = src[(size_t)c * N + n];
    if (c + 8 < K) __builtin_prefetch(&src[(size_t)(c + 8) * N + n], 0, 0);
  }
  bf16 h = (bf16)v;
  hi[idx] = h;
  lo[idx] = (bf16)(v - (float)h);
}

// adj = rownorm( sign(G) + I ), in place. One block per row.
__global__ __launch_bounds__(256) void build_adj_inplace(float* __restrict__ G, int n)
{
  __shared__ float red[256];
  const int i = blockIdx.x;
  const int t = threadIdx.x;
  float s = 0.0f;
  for (int j = t; j < n; j += 256) {
    float g = G[(size_t)i * n + j];
    float a = (g > 0.0f) ? 1.0f : ((g < 0.0f) ? -1.0f : 0.0f);
    if (j == i) a += 1.0f;
    s += a;
  }
  red[t] = s;
  __syncthreads();
  for (int w = 128; w > 0; w >>= 1) {
    if (t < w) red[t] += red[t + w];
    __syncthreads();
  }
  const float rs = red[0];
  const float rinv = (rs == 0.0f) ? 0.0f : (1.0f / rs);
  for (int j = t; j < n; j += 256) {
    float g = G[(size_t)i * n + j];
    float a = (g > 0.0f) ? 1.0f : ((g < 0.0f) ? -1.0f : 0.0f);
    if (j == i) a += 1.0f;
    G[(size_t)i * n + j] = a * rinv;
  }
}

// One k-step of fragment loads + split-precision WMMA for a 64x32 wave tile.
__device__ __forceinline__ void compute_step(
    const bf16* pAhi, const bf16* pAlo, const bf16* pBhi, const bf16* pBlo,
    int wm, int wn, int lr, int lh, v8f (&acc)[4][2])
{
  Frag ah[4], al[4], bh[2], bl[2];
#pragma unroll
  for (int i = 0; i < 4; ++i) {
    const int row = (wm + i * 16 + lr) * BK;
    ah[i].q[0] = *(const uint4*)&pAhi[row + lh * 8];
    ah[i].q[1] = *(const uint4*)&pAhi[row + 16 + lh * 8];
    al[i].q[0] = *(const uint4*)&pAlo[row + lh * 8];
    al[i].q[1] = *(const uint4*)&pAlo[row + 16 + lh * 8];
  }
#pragma unroll
  for (int j = 0; j < 2; ++j) {
    const int row = (wn + j * 16 + lr) * BK;
    bh[j].q[0] = *(const uint4*)&pBhi[row + lh * 8];
    bh[j].q[1] = *(const uint4*)&pBhi[row + 16 + lh * 8];
    bl[j].q[0] = *(const uint4*)&pBlo[row + lh * 8];
    bl[j].q[1] = *(const uint4*)&pBlo[row + 16 + lh * 8];
  }
  // Split-precision accumulate: hi*hi + hi*lo + lo*hi.
#pragma unroll
  for (int i = 0; i < 4; ++i)
#pragma unroll
    for (int j = 0; j < 2; ++j) {
      acc[i][j] = __builtin_amdgcn_wmma_f32_16x16x32_bf16(
          false, ah[i].v, false, bh[j].v, (short)0, acc[i][j], false, false);
      acc[i][j] = __builtin_amdgcn_wmma_f32_16x16x32_bf16(
          false, ah[i].v, false, bl[j].v, (short)0, acc[i][j], false, false);
      acc[i][j] = __builtin_amdgcn_wmma_f32_16x16x32_bf16(
          false, al[i].v, false, bh[j].v, (short)0, acc[i][j], false, false);
    }
}

// C = A * B (+bias, +act) with split-bf16 operands.
// A: M x Kp row-major (hi/lo). BT: N x Kp row-major (i.e. B transposed, K-major).
// act: 0 = none, 1 = relu, 2 = write C0 = v and C1 = tanh(v).
// M % 128 == 0; Kp % 32 == 0; N % 16 == 0 (partial BN tiles guarded).
__global__ __launch_bounds__(256) void gemm_bf16split(
    const bf16* __restrict__ Ahi, const bf16* __restrict__ Alo,
    const bf16* __restrict__ BThi, const bf16* __restrict__ BTlo,
    const float* __restrict__ bias,
    float* __restrict__ C0, float* __restrict__ C1,
    int M, int N, int Kp, int act)
{
  const int t  = threadIdx.x;
  const int m0 = blockIdx.y * BM;
  const int n0 = blockIdx.x * BN;
  const int lane = t & 31;
  const int wv = t >> 5;          // wave 0..7
  const int wm = (wv & 1) * 64;   // wave M offset in block tile
  const int wn = (wv >> 1) * 32;  // wave N offset in block tile
  const int lr = lane & 15;       // row/col within 16x16 fragment
  const int lh = lane >> 4;       // K-half selector per ISA layout

  v8f acc[4][2];
#pragma unroll
  for (int i = 0; i < 4; ++i)
#pragma unroll
    for (int j = 0; j < 2; ++j)
      acc[i][j] = (v8f){0, 0, 0, 0, 0, 0, 0, 0};

  const int ksteps = Kp / BK;

#ifdef GCN_ASYNC
  // Double-buffered LDS, filled by ASYNCcnt-tracked global_load_async_to_lds.
  __shared__ __align__(16) bf16 sA[2][2][BM * BK];  // [buf][hi/lo]
  __shared__ __align__(16) bf16 sB[2][2][BN * BK];

  // Zero out-of-range B rows once; async loads never touch them, so the zeros
  // persist across all k-steps in both ping-pong buffers.
  if (n0 + BN > N) {
    for (int u = t; u < 512; u += 256) {
      const int r = u >> 2;
      const int c = (u & 3) << 3;
      if (n0 + r >= N) {
        const uint4 z = make_uint4(0, 0, 0, 0);
        *(uint4*)&sB[0][0][r * BK + c] = z;
        *(uint4*)&sB[0][1][r * BK + c] = z;
        *(uint4*)&sB[1][0][r * BK + c] = z;
        *(uint4*)&sB[1][1][r * BK + c] = z;
      }
    }
  }
  __syncthreads();

  auto stage = [&](int kk, int buf) {
#pragma unroll
    for (int u = t; u < 512; u += 256) {
      const int r = u >> 2;
      const int c = (u & 3) << 3;
      const int ga = (m0 + r) * Kp + kk * BK + c;
      async_cp16(Ahi + ga, &sA[buf][0][r * BK + c]);
      async_cp16(Alo + ga, &sA[buf][1][r * BK + c]);
      if (n0 + r < N) {
        const int gb = (n0 + r) * Kp + kk * BK + c;
        async_cp16(BThi + gb, &sB[buf][0][r * BK + c]);
        async_cp16(BTlo + gb, &sB[buf][1][r * BK + c]);
      }
    }
  };

  stage(0, 0);
  wait_async0();
  __syncthreads();

  for (int kk = 0; kk < ksteps; ++kk) {
    const int cur = kk & 1;
    if (kk + 1 < ksteps) stage(kk + 1, cur ^ 1);  // overlap with WMMA below
    compute_step(sA[cur][0], sA[cur][1], sB[cur][0], sB[cur][1],
                 wm, wn, lr, lh, acc);
    wait_async0();    // own wave's async writes to the other buffer are done
    __syncthreads();  // everyone finished reading cur + finished their writes
  }
#else
  // Fallback: synchronous register-staged LDS tiles.
  __shared__ __align__(16) bf16 sAhi[BM * BK];
  __shared__ __align__(16) bf16 sAlo[BM * BK];
  __shared__ __align__(16) bf16 sBhi[BN * BK];
  __shared__ __align__(16) bf16 sBlo[BN * BK];

  for (int kk = 0; kk < ksteps; ++kk) {
    const int k0 = kk * BK;
    __syncthreads();
#pragma unroll
    for (int u = t; u < 512; u += 256) {
      const int r = u >> 2;
      const int c = (u & 3) << 3;
      const int g = (m0 + r) * Kp + k0 + c;
      *(uint4*)&sAhi[r * BK + c] = *(const uint4*)&Ahi[g];
      *(uint4*)&sAlo[r * BK + c] = *(const uint4*)&Alo[g];
    }
#pragma unroll
    for (int u = t; u < 512; u += 256) {
      const int r = u >> 2;
      const int c = (u & 3) << 3;
      uint4 h4 = make_uint4(0, 0, 0, 0);
      uint4 l4 = make_uint4(0, 0, 0, 0);
      if (n0 + r < N) {
        const int g = (n0 + r) * Kp + k0 + c;
        h4 = *(const uint4*)&BThi[g];
        l4 = *(const uint4*)&BTlo[g];
      }
      *(uint4*)&sBhi[r * BK + c] = h4;
      *(uint4*)&sBlo[r * BK + c] = l4;
    }
    __syncthreads();
    compute_step(sAhi, sAlo, sBhi, sBlo, wm, wn, lr, lh, acc);
  }
#endif

  // Epilogue: C/D lane layout — col = lane&15, row = 8*(lane>>4) + vgpr.
#pragma unroll
  for (int i = 0; i < 4; ++i) {
#pragma unroll
    for (int j = 0; j < 2; ++j) {
      const int gn = n0 + wn + j * 16 + lr;
      if (gn < N) {
        const float bv = bias ? bias[gn] : 0.0f;
#pragma unroll
        for (int r = 0; r < 8; ++r) {
          const int gm = m0 + wm + i * 16 + lh * 8 + r;
          float v = acc[i][j][r] + bv;
          if (act == 1) v = fmaxf(v, 0.0f);
          const size_t o = (size_t)gm * N + gn;
          if (act == 2) { C0[o] = v; C1[o] = tanhf(v); }
          else          { C0[o] = v; }
        }
      }
    }
  }
}

extern "C" void kernel_launch(void* const* d_in, const int* in_sizes, int n_in,
                              void* d_out, int out_size, void* d_ws, size_t ws_size,
                              hipStream_t stream)
{
  const float* x  = (const float*)d_in[0];
  const float* W1 = (const float*)d_in[1];
  const float* b1 = (const float*)d_in[2];
  const float* W2 = (const float*)d_in[3];
  const float* b2 = (const float*)d_in[4];
  const float* W3 = (const float*)d_in[5];
  const float* b3 = (const float*)d_in[6];

  const int N = 2048, K1 = 1386, K1p = 1408, H = 4096, C = 64;

  char* ws = (char*)d_ws;
  size_t off = 0;
  auto alloc = [&](size_t bytes) -> char* {
    char* p = ws + off;
    off += (bytes + 255) & ~(size_t)255;
    return p;
  };

  float* G  = (float*)alloc((size_t)N * N * 4);   // G, then adj (in place)
  float* P  = (float*)alloc((size_t)N * H * 4);   // H1 -> F1 -> T2 -> T3
  bf16* xhi = (bf16*)alloc((size_t)N * K1p * 2);
  bf16* xlo = (bf16*)alloc((size_t)N * K1p * 2);
  bf16* ahi = (bf16*)alloc((size_t)N * N * 2);    // adj split (persistent)
  bf16* alo = (bf16*)alloc((size_t)N * N * 2);
  bf16* whi = (bf16*)alloc((size_t)H * H * 2);    // W1^T then W2^T
  bf16* wlo = (bf16*)alloc((size_t)H * H * 2);
  bf16* rhi = (bf16*)alloc((size_t)H * N * 2);    // rotating: H1^T/F1/T2^T/feat
  bf16* rlo = (bf16*)alloc((size_t)H * N * 2);
  bf16* w3hi = (bf16*)alloc((size_t)C * H * 2);
  bf16* w3lo = (bf16*)alloc((size_t)C * H * 2);
  bf16* t3hi = (bf16*)alloc((size_t)C * N * 2);
  bf16* t3lo = (bf16*)alloc((size_t)C * N * 2);

  float* feat = (float*)d_out;                    // 2048 x 4096
  float* hid  = feat + (size_t)N * H;             // 2048 x 64
  float* code = hid + (size_t)N * C;              // 2048 x 64

  dim3 blk(256);
  auto cdiv = [](int a, int b) { return (a + b - 1) / b; };

  // 1) x -> split bf16 (padded K)
  cvt_split_rowmajor<<<cdiv(N * K1p, 256), blk, 0, stream>>>(x, xhi, xlo, N, K1, K1p);
  // 2) G = x @ x^T  (x serves as both A and BT since BT[n][k] = x[n][k])
  gemm_bf16split<<<dim3(N / BN, N / BM), blk, 0, stream>>>(
      xhi, xlo, xhi, xlo, nullptr, G, nullptr, N, N, K1p, 0);
  // 3) adjacency + row-normalize, in place
  build_adj_inplace<<<N, blk, 0, stream>>>(G, N);
  // 4) adj -> split bf16 (A operand)
  cvt_split_rowmajor<<<cdiv(N * N, 256), blk, 0, stream>>>(G, ahi, alo, N, N, N);
  // 5) W1^T -> split bf16
  cvt_split_transpose<<<cdiv(H * K1p, 256), blk, 0, stream>>>(W1, whi, wlo, K1, H, K1p);
  // 6) H1 = x @ W1 -> P
  gemm_bf16split<<<dim3(H / BN, N / BM), blk, 0, stream>>>(
      xhi, xlo, whi, wlo, nullptr, P, nullptr, N, H, K1p, 0);
  // 7) H1^T -> split bf16 (rotating)
  cvt_split_transpose<<<cdiv(H * N, 256), blk, 0, stream>>>(P, rhi, rlo, N, H, N);
  // 8) F1 = adj @ H1 + b1 -> P
  gemm_bf16split<<<dim3(H / BN, N / BM), blk, 0, stream>>>(
      ahi, alo, rhi, rlo, b1, P, nullptr, N, H, N, 0);
  // 9) F1 -> split bf16 (A operand, rotating)
  cvt_split_rowmajor<<<cdiv(N * H, 256), blk, 0, stream>>>(P, rhi, rlo, N, H, H);
  // 10) W2^T -> split bf16
  cvt_split_transpose<<<cdiv(H * H, 256), blk, 0, stream>>>(W2, whi, wlo, H, H, H);
  // 11) T2 = F1 @ W2 -> P
  gemm_bf16split<<<dim3(H / BN, N / BM), blk, 0, stream>>>(
      rhi, rlo, whi, wlo, nullptr, P, nullptr, N, H, H, 0);
  // 12) T2^T -> split bf16 (rotating)
  cvt_split_transpose<<<cdiv(H * N, 256), blk, 0, stream>>>(P, rhi, rlo, N, H, N);
  // 13) feat = relu(adj @ T2 + b2) -> d_out
  gemm_bf16split<<<dim3(H / BN, N / BM), blk, 0, stream>>>(
      ahi, alo, rhi, rlo, b2, feat, nullptr, N, H, N, 1);
  // 14) feat -> split bf16 (A operand, rotating)
  cvt_split_rowmajor<<<cdiv(N * H, 256), blk, 0, stream>>>(feat, rhi, rlo, N, H, H);
  // 15) W3^T -> split bf16
  cvt_split_transpose<<<cdiv(C * H, 256), blk, 0, stream>>>(W3, w3hi, w3lo, H, C, H);
  // 16) T3 = feat @ W3 -> P
  gemm_bf16split<<<dim3(cdiv(C, BN), N / BM), blk, 0, stream>>>(
      rhi, rlo, w3hi, w3lo, nullptr, P, nullptr, N, C, H, 0);
  // 17) T3^T -> split bf16
  cvt_split_transpose<<<cdiv(C * N, 256), blk, 0, stream>>>(P, t3hi, t3lo, N, C, N);
  // 18) hid = adj @ T3 + b3 ; code = tanh(hid)
  gemm_bf16split<<<dim3(cdiv(C, BN), N / BM), blk, 0, stream>>>(
      ahi, alo, t3hi, t3lo, b3, hid, code, N, C, N, 2);

  (void)in_sizes; (void)n_in; (void)out_size; (void)ws_size;
}
